// MXIntMatMul_39797166964937
// MI455X (gfx1250) — matrix operand order repaired and checked by reading, not compiled, not run on hardware
//
#include <hip/hip_runtime.h>

typedef __attribute__((ext_vector_type(16))) _Float16 v16h;
typedef __attribute__((ext_vector_type(8)))  float    v8f;
typedef __attribute__((ext_vector_type(4)))  float    v4f;

union FragAB { v16h h; v4f f[2]; };

// ---- MX quant helpers (width=8, exp_width=8) -------------------------------
// e = clamp(ceil(log2(max(|block|, 2^-126))), -128, 127), exact via frexpf.
__device__ __forceinline__ int mx_exp(float blockmax) {
  float m = fmaxf(blockmax, 1.1754943508222875e-38f); // 2^-126
  int e;
  float f = frexpf(m, &e);        // m = f * 2^e, f in [0.5, 1)
  if (f == 0.5f) e -= 1;          // exact power of two: ceil(log2) = e-1
  return e < -128 ? -128 : (e > 127 ? 127 : e);
}

// q = clamp(rint(v * 2^(7-e)), -128, 127) * 2^(e-7); rintf == round-half-even.
__device__ __forceinline__ float mx_apply(float v, int e) {
  float mant = rintf(ldexpf(v, 7 - e));
  mant = fminf(fmaxf(mant, -128.0f), 127.0f);
  return ldexpf(mant, e - 7);
}

__device__ __forceinline__ float wave32_block_absmax(float v) {
  float m = fabsf(v);
#pragma unroll
  for (int off = 16; off > 0; off >>= 1)
    m = fmaxf(m, __shfl_xor(m, off, 32));
  return m;
}

// ---- Kernel 1: quantize x row-blocks (1x32 along last dim) -> f16 ----------
// One wave32 == one MX block; fully coalesced.
__global__ void __launch_bounds__(256) mx_quant_rows_f16(
    const float* __restrict__ in, _Float16* __restrict__ out) {
  size_t i = (size_t)blockIdx.x * blockDim.x + threadIdx.x;
  float v = in[i];
  int e = mx_exp(wave32_block_absmax(v));
  out[i] = (_Float16)mx_apply(v, e);
}

// ---- Kernel 2: quantize y (blocks 1x32 along N) and store transposed -------
// qyT[n][k] = quant(y[k][n]); coalesced loads, strided f16 stores (y is tiny).
__global__ void __launch_bounds__(256) mx_quant_y_t(
    const float* __restrict__ y, _Float16* __restrict__ qyT, int K, int N) {
  int lane = threadIdx.x & 31;
  int w = blockIdx.x * (blockDim.x >> 5) + (threadIdx.x >> 5);
  int nb = N >> 5;
  int k = w / nb;
  int n = ((w % nb) << 5) + lane;
  float v = y[(size_t)k * N + n];
  int e = mx_exp(wave32_block_absmax(v));
  qyT[(size_t)n * K + k] = (_Float16)mx_apply(v, e);
}

// ---- Kernel 3: WMMA f16 GEMM with fused MX requant epilogue ----------------
// Block: 128 threads = 4 waves (2x2), block tile 128x128, wave tile 64x64
// (4x4 of 16x16x32 WMMA -> 16 WMMAs per 16 b128 loads per K-step).
// A: [M][K] row-major f16. Bt: [N][K] row-major f16.
// A fragment layout (ISA 16-bit A 16x32): lane L holds M=L%16,
//   K = (L<16 ? 0 : 8) + {0..7} and +16 + {0..7}  -> two b128 loads.
// B fragment layout (ISA 16-bit B): lane L holds N=L%16,
//   K = (L<16 ? 0 : 8) + {0..7} and +16 + {0..7}  -> two b128 loads.
__global__ void __launch_bounds__(128) mx_gemm_wmma(
    const _Float16* __restrict__ A, const _Float16* __restrict__ Bt,
    float* __restrict__ C, int M, int N, int K) {
  const int lane = threadIdx.x & 31;
  const int wid  = threadIdx.x >> 5;
  const int wm = wid >> 1, wn = wid & 1;
  const int m0 = blockIdx.y * 128 + wm * 64;
  const int n0 = blockIdx.x * 128 + wn * 64;
  const int lr  = lane & 15;
  const int hi  = lane >> 4;       // half-wave select
  const int ka  = hi * 8;          // A per-lane K base offset
  const int kbo = hi * 8;          // B per-lane K base offset

  const _Float16* ap[4];
  const _Float16* bp[4];
#pragma unroll
  for (int t = 0; t < 4; ++t) {
    ap[t] = A  + (size_t)(m0 + 16 * t + lr) * K + ka;
    bp[t] = Bt + (size_t)(n0 + 16 * t + lr) * K + kbo;
  }

  v8f acc[4][4] = {};

  for (int k = 0; k < K; k += 32) {
    FragAB fa[4], fb[4];
#pragma unroll
    for (int t = 0; t < 4; ++t) {
      fa[t].f[0] = *(const v4f*)(ap[t] + k);
      fa[t].f[1] = *(const v4f*)(ap[t] + k + 16);
    }
#pragma unroll
    for (int t = 0; t < 4; ++t) {
      fb[t].f[0] = *(const v4f*)(bp[t] + k);
      fb[t].f[1] = *(const v4f*)(bp[t] + k + 16);
    }
#pragma unroll
    for (int i = 0; i < 4; ++i)
#pragma unroll
      for (int j = 0; j < 4; ++j)
        acc[i][j] = __builtin_amdgcn_wmma_f32_16x16x32_f16(
            false, fa[i].h, false, fb[j].h, (short)0, acc[i][j], false, false);
  }

  // Fused output MX requant: each output row's 32-wide N block lives in
  // acc[i][2*jp][r] (cols +0..15) and acc[i][2*jp+1][r] (cols +16..31) across
  // a 16-lane half-wave; xor-shuffles (1,2,4,8) stay within the half-wave.
#pragma unroll
  for (int i = 0; i < 4; ++i) {
#pragma unroll
    for (int jp = 0; jp < 2; ++jp) {
#pragma unroll
      for (int r = 0; r < 8; ++r) {
        float va = acc[i][2 * jp][r];
        float vb = acc[i][2 * jp + 1][r];
        float m = fmaxf(fabsf(va), fabsf(vb));
#pragma unroll
        for (int off = 1; off < 16; off <<= 1)
          m = fmaxf(m, __shfl_xor(m, off, 32));
        int e = mx_exp(m);
        size_t row = (size_t)(m0 + 16 * i + hi * 8 + r);
        float* cp = C + row * N + n0 + 32 * jp + lr;
        cp[0]  = mx_apply(va, e);
        cp[16] = mx_apply(vb, e);
      }
    }
  }
}

extern "C" void kernel_launch(void* const* d_in, const int* in_sizes, int n_in,
                              void* d_out, int out_size, void* d_ws, size_t ws_size,
                              hipStream_t stream) {
  const float* x = (const float*)d_in[0];   // [4,4096,1024] -> [16384,1024]
  const float* y = (const float*)d_in[1];   // [1024,1024]
  float* out = (float*)d_out;               // [16384,1024]

  const int K = 1024;
  const int N = 1024;
  const int M = in_sizes[0] / K;            // 16384

  _Float16* qx  = (_Float16*)d_ws;                                   // 32 MB
  _Float16* qyT = (_Float16*)((char*)d_ws + (size_t)M * K * sizeof(_Float16)); // 2 MB

  // 1) quantize x -> f16 (exact MXINT8 values)
  {
    size_t n = (size_t)M * K;
    mx_quant_rows_f16<<<dim3((unsigned)(n / 256)), dim3(256), 0, stream>>>(x, qx);
  }
  // 2) quantize y -> f16, stored transposed [N][K]
  {
    int waves = K * (N / 32);
    mx_quant_y_t<<<dim3(waves * 32 / 256), dim3(256), 0, stream>>>(y, qyT, K, N);
  }
  // 3) WMMA GEMM (64x64 per wave) + fused output requant
  mx_gemm_wmma<<<dim3(N / 128, M / 128), dim3(128), 0, stream>>>(qx, qyT, out, M, N, K);
}
